// LearnedConv_63376537420170
// MI455X (gfx1250) — matrix operand order repaired and checked by reading, not compiled, not run on hardware
//
#include <hip/hip_runtime.h>
#include <hip/hip_bf16.h>

typedef __attribute__((ext_vector_type(16))) _Float16 v16h;
typedef __attribute__((ext_vector_type(8)))  _Float16 v8h;
typedef __attribute__((ext_vector_type(8)))  float    v8f;

#define N_NODES   100000
#define N_EDGES   1600000
#define N_TILES   (N_NODES / 16)   // 6250, exact
#define HDIM      16

// ---------------------------------------------------------------------------
// Kernel 0: zero accumulators (node max keys | denominators | h accumulator)
// ---------------------------------------------------------------------------
__global__ __launch_bounds__(256) void k_init(unsigned* zbase, int n_dwords) {
    int stride = gridDim.x * blockDim.x;
    for (int i = blockIdx.x * blockDim.x + threadIdx.x; i < n_dwords; i += stride)
        zbase[i] = 0u;
}

// ---------------------------------------------------------------------------
// Kernel 0b: one-shot conversion of Wq|Wk|Wv|Wo (4 x 64x64 f32) to f16
// ---------------------------------------------------------------------------
__global__ __launch_bounds__(256) void k_wconv(const float* __restrict__ Wq,
                                               const float* __restrict__ Wk,
                                               const float* __restrict__ Wv,
                                               const float* __restrict__ Wo,
                                               _Float16* __restrict__ Wh) {
    int i = blockIdx.x * 256 + threadIdx.x;
    if (i >= 4 * 4096) return;
    int m = i >> 12;
    int j = i & 4095;
    const float* W = (m == 0) ? Wq : (m == 1) ? Wk : (m == 2) ? Wv : Wo;
    Wh[i] = (_Float16)W[j];
}

// ---------------------------------------------------------------------------
// Kernel 1: Q/K/V projection via v_wmma_f32_16x16x32_f16, f16 output.
// blockIdx.y = matrix m in {Q,K,V}; the block stages W_m (f16, 8KB) in LDS
// once, then its 8 waves each compute one 16x64 tile (8 WMMA each).
// ---------------------------------------------------------------------------
__global__ __launch_bounds__(256) void k_qkv(const float* __restrict__ src_feat,
                                             const float* __restrict__ dst_feat,
                                             const _Float16* __restrict__ Wh,
                                             _Float16* __restrict__ qkv_h) {
    __shared__ _Float16 Wl[4096];                 // 8 KB
    int m = blockIdx.y;
    // stage W_m into LDS (2048 dwords, 8 per thread)
    {
        const unsigned* wsrc = (const unsigned*)(Wh + m * 4096);
        unsigned* wdst = (unsigned*)Wl;
        #pragma unroll
        for (int i = threadIdx.x; i < 2048; i += 256) wdst[i] = wsrc[i];
    }
    __syncthreads();

    int tile = blockIdx.x * 8 + (threadIdx.x >> 5);
    if (tile >= N_TILES) return;

    const float* X = (m == 0) ? dst_feat : src_feat;   // Q uses dst_feat
    _Float16* Y = qkv_h + (size_t)m * N_NODES * 64;

    int lane = threadIdx.x & 31;
    int r    = lane & 15;       // A row / B-D column within tile
    int half = lane >> 4;       // lane-group selector
    int row  = tile * 16 + r;

    v8f acc0 = {}, acc1 = {}, acc2 = {}, acc3 = {};
    #pragma unroll
    for (int k0 = 0; k0 < 64; k0 += 32) {
        // ---- A operand: X[16 rows, k0..k0+31], f32 -> f16, 16-bit A layout ----
        v16h a;
        const float* xp = X + (size_t)row * 64 + k0 + half * 8;
        #pragma unroll
        for (int j = 0; j < 8; ++j) {
            a[j]     = (_Float16)xp[j];        // K = k0 + half*8 + j
            a[8 + j] = (_Float16)xp[16 + j];   // K = k0 + half*8 + 16 + j
        }
        // ---- B operands from LDS: B[k][n] = W[n][k], 4 column tiles ----
        #pragma unroll
        for (int ct = 0; ct < 4; ++ct) {
            const v8h* bp = (const v8h*)(Wl + (ct * 16 + r) * 64 + k0 + half * 8);
            v8h blo = bp[0], bhi = bp[2];      // ds_load_b128 x2
            v16h b;
            #pragma unroll
            for (int j = 0; j < 8; ++j) { b[j] = blo[j]; b[8 + j] = bhi[j]; }
            v8f& acc = (ct == 0) ? acc0 : (ct == 1) ? acc1 : (ct == 2) ? acc2 : acc3;
            acc = __builtin_amdgcn_wmma_f32_16x16x32_f16(
                      false, a, false, b, (short)0, acc, false, false);
        }
    }
    // ---- store D as f16: lane holds column r of rows half*8 + v ----
    #pragma unroll
    for (int ct = 0; ct < 4; ++ct) {
        const v8f& acc = (ct == 0) ? acc0 : (ct == 1) ? acc1 : (ct == 2) ? acc2 : acc3;
        #pragma unroll
        for (int v = 0; v < 8; ++v)
            Y[(size_t)(tile * 16 + half * 8 + v) * 64 + ct * 16 + r] = (_Float16)acc[v];
    }
}

// ---------------------------------------------------------------------------
// Kernel 2: per-(edge, head) attention score + monotone-key atomic segment max
// ---------------------------------------------------------------------------
__global__ __launch_bounds__(256) void k_scores(const _Float16* __restrict__ Qh,
                                                const _Float16* __restrict__ Kh,
                                                const float* __restrict__ edge_feat,
                                                const int* __restrict__ src_idx,
                                                const int* __restrict__ dst_idx,
                                                const float* __restrict__ We,
                                                const float* __restrict__ be,
                                                float* __restrict__ scores,
                                                unsigned* __restrict__ nmax) {
    int gid = blockIdx.x * 256 + threadIdx.x;
    int e = gid >> 2;
    if (e >= N_EDGES) return;
    int h = gid & 3;
    int s = src_idx[e];
    int d = dst_idx[e];

    const v8h* qp = (const v8h*)(Qh + (size_t)d * 64 + h * HDIM);
    const v8h* kp = (const v8h*)(Kh + (size_t)s * 64 + h * HDIM);
    v8h q0 = qp[0], q1 = qp[1];
    v8h k0 = kp[0], k1 = kp[1];
    float dot = 0.f;
    #pragma unroll
    for (int i = 0; i < 8; ++i)
        dot += (float)q0[i] * (float)k0[i] + (float)q1[i] * (float)k1[i];
    float sc = dot * 0.25f;                      // HEAD_DIM^-0.5 = 1/4

    const float* ef = edge_feat + (size_t)e * 16;
    const float* we = We + h * 16;
    float eb = be[h];
    #pragma unroll
    for (int i = 0; i < 16; ++i) eb += ef[i] * we[i];
    sc += eb;

    scores[(size_t)e * 4 + h] = sc;
    unsigned bits = __float_as_uint(sc);
    unsigned key  = (bits & 0x80000000u) ? ~bits : (bits | 0x80000000u);
    atomicMax(nmax + (size_t)d * 4 + h, key);
}

// ---------------------------------------------------------------------------
// Kernel 3: per-(edge, feature) exp + scatter-add of numerator and denominator
// ---------------------------------------------------------------------------
__global__ __launch_bounds__(256) void k_accum(const _Float16* __restrict__ Vh,
                                               const float* __restrict__ scores,
                                               const unsigned* __restrict__ nmax,
                                               const int* __restrict__ src_idx,
                                               const int* __restrict__ dst_idx,
                                               float* __restrict__ den,
                                               float* __restrict__ hacc) {
    long long gid = (long long)blockIdx.x * 256 + threadIdx.x;
    int e = (int)(gid >> 6);
    if (e >= N_EDGES) return;
    int t = (int)(gid & 63);
    int h = t >> 4;
    int d = dst_idx[e];
    int s = src_idx[e];

    float sc = scores[(size_t)e * 4 + h];
    unsigned key = nmax[(size_t)d * 4 + h];
    unsigned mb  = (key & 0x80000000u) ? (key ^ 0x80000000u) : ~key;
    float m = __uint_as_float(mb);
    float ex = __expf(sc - m);

    if ((t & 15) == 0) atomicAdd(den + (size_t)d * 4 + h, ex);
    float v = (float)Vh[(size_t)s * 64 + t];
    atomicAdd(hacc + (size_t)d * 64 + t, ex * v);
}

// ---------------------------------------------------------------------------
// Kernel 4: normalize by denominator + output projection (WMMA) + bias.
// Wo (f16) staged in LDS per block; 8 waves per block, one 16-node tile each.
// ---------------------------------------------------------------------------
__global__ __launch_bounds__(256) void k_out(const float* __restrict__ hacc,
                                             const float* __restrict__ den,
                                             const _Float16* __restrict__ Wh,
                                             const float* __restrict__ bo,
                                             float* __restrict__ out) {
    __shared__ _Float16 Wl[4096];                 // Wo in f16, 8 KB
    {
        const unsigned* wsrc = (const unsigned*)(Wh + 3 * 4096);
        unsigned* wdst = (unsigned*)Wl;
        #pragma unroll
        for (int i = threadIdx.x; i < 2048; i += 256) wdst[i] = wsrc[i];
    }
    __syncthreads();

    int tile = blockIdx.x * 8 + (threadIdx.x >> 5);
    if (tile >= N_TILES) return;
    int lane = threadIdx.x & 31;
    int r    = lane & 15;
    int half = lane >> 4;
    int row  = tile * 16 + r;

    v8f acc0 = {}, acc1 = {}, acc2 = {}, acc3 = {};
    #pragma unroll
    for (int k0 = 0; k0 < 64; k0 += 32) {
        int kbase = k0 + half * 8;              // 0, 8, 32, 40
        int h0 = kbase >> 4;
        int h1 = (kbase + 16) >> 4;
        float dn0 = den[(size_t)row * 4 + h0];
        float dn1 = den[(size_t)row * 4 + h1];
        float iv0 = dn0 > 0.f ? 1.0f / dn0 : 0.f;   // no-in-edge nodes -> h = 0
        float iv1 = dn1 > 0.f ? 1.0f / dn1 : 0.f;
        const float* hp = hacc + (size_t)row * 64 + kbase;
        v16h a;
        #pragma unroll
        for (int j = 0; j < 8; ++j) {
            a[j]     = (_Float16)(hp[j] * iv0);
            a[8 + j] = (_Float16)(hp[16 + j] * iv1);
        }
        #pragma unroll
        for (int ct = 0; ct < 4; ++ct) {
            const v8h* bp = (const v8h*)(Wl + (ct * 16 + r) * 64 + k0 + half * 8);
            v8h blo = bp[0], bhi = bp[2];
            v16h b;
            #pragma unroll
            for (int j = 0; j < 8; ++j) { b[j] = blo[j]; b[8 + j] = bhi[j]; }
            v8f& acc = (ct == 0) ? acc0 : (ct == 1) ? acc1 : (ct == 2) ? acc2 : acc3;
            acc = __builtin_amdgcn_wmma_f32_16x16x32_f16(
                      false, a, false, b, (short)0, acc, false, false);
        }
    }
    #pragma unroll
    for (int ct = 0; ct < 4; ++ct) {
        const v8f& acc = (ct == 0) ? acc0 : (ct == 1) ? acc1 : (ct == 2) ? acc2 : acc3;
        float bias = bo[ct * 16 + r];
        #pragma unroll
        for (int v = 0; v < 8; ++v)
            out[(size_t)(tile * 16 + half * 8 + v) * 64 + ct * 16 + r] = acc[v] + bias;
    }
}

// ---------------------------------------------------------------------------
// Host-side orchestration (graph-capture safe: kernels only, all on `stream`)
// ---------------------------------------------------------------------------
extern "C" void kernel_launch(void* const* d_in, const int* in_sizes, int n_in,
                              void* d_out, int out_size, void* d_ws, size_t ws_size,
                              hipStream_t stream) {
    const float* src_feat  = (const float*)d_in[0];
    const float* dst_feat  = (const float*)d_in[1];
    const float* edge_feat = (const float*)d_in[2];
    const int*   src_idx   = (const int*)d_in[3];
    const int*   dst_idx   = (const int*)d_in[4];
    const float* Wq        = (const float*)d_in[5];
    const float* Wk        = (const float*)d_in[6];
    const float* Wv        = (const float*)d_in[7];
    const float* We        = (const float*)d_in[8];
    const float* be        = (const float*)d_in[9];
    const float* Wo        = (const float*)d_in[10];
    const float* bo        = (const float*)d_in[11];
    float* out = (float*)d_out;

    // workspace layout
    _Float16* qkv_h = (_Float16*)d_ws;                       // 3 * N * 64 halves
    _Float16* Qh    = qkv_h;
    _Float16* Kh    = qkv_h + (size_t)N_NODES * 64;
    _Float16* Vh    = qkv_h + (size_t)2 * N_NODES * 64;
    _Float16* Wh    = qkv_h + (size_t)3 * N_NODES * 64;      // 4 * 4096 halves
    float*    scores= (float*)(Wh + 4 * 4096);               // E * 4 f32
    unsigned* nmax  = (unsigned*)(scores + (size_t)N_EDGES * 4); // N * 4
    float*    den   = (float*)(nmax + (size_t)N_NODES * 4);  // N * 4
    float*    hacc  = den + (size_t)N_NODES * 4;             // N * 64 f32

    // 0) zero nmax | den | hacc (contiguous): N*(4+4+64) dwords
    k_init<<<4096, 256, 0, stream>>>(nmax, N_NODES * 72);

    // 0b) convert weights to f16 once
    k_wconv<<<64, 256, 0, stream>>>(Wq, Wk, Wv, Wo, Wh);

    // 1) Q/K/V projections: grid (782 blocks x 3 matrices), 8 waves per block
    dim3 qkv_grid((N_TILES + 7) / 8, 3);
    k_qkv<<<qkv_grid, 256, 0, stream>>>(src_feat, dst_feat, Wh, qkv_h);

    // 2) edge scores + segment max: E*4 threads
    k_scores<<<(N_EDGES * 4 + 255) / 256, 256, 0, stream>>>(
        Qh, Kh, edge_feat, src_idx, dst_idx, We, be, scores, nmax);

    // 3) exp + scatter accumulate: E*64 threads
    long long acc_threads = (long long)N_EDGES * 64;
    k_accum<<<(int)((acc_threads + 255) / 256), 256, 0, stream>>>(
        Vh, scores, nmax, src_idx, dst_idx, den, hacc);

    // 4) normalize + output projection + bias: 6250 waves
    k_out<<<(N_TILES + 7) / 8, 256, 0, stream>>>(hacc, den, Wh, bo, out);
}
